// BigramLanguageModel_67044439490742
// MI455X (gfx1250) — matrix operand order repaired and checked by reading, not compiled, not run on hardware
//
#include <hip/hip_runtime.h>

typedef _Float16 half_t;
typedef __attribute__((ext_vector_type(16))) _Float16 v16h;
typedef __attribute__((ext_vector_type(8)))  _Float16 v8h;
typedef __attribute__((ext_vector_type(8)))  float    v8f;

#define Bc  2
#define Tc  1024
#define Cc  1024
#define Hc  16
#define HSc 64
#define Lc  6
#define Vc  32000
#define FFc 4096
#define MR  (Bc*Tc)           // 2048 rows
#define SCALE_F 5.656854249492380f  // 1024^0.25 (faithful to reference)

// raw split workgroup barrier: no implicit vmem-drain fence (the async-copy
// producer side is ordered by s_wait_asynccnt; the LDS consumer side retires
// its ds_loads before the preceding WMMAs). Keeps B-fragment global loads in
// flight across the barrier.
#define WG_BARRIER() asm volatile("s_barrier_signal -1\ns_barrier_wait -1" ::: "memory")

union frag {
    v16h v;
    v8h  h[2];
};

// ---------------------------------------------------------------- embedding
__global__ void k_embed(const int* __restrict__ idx, const float* __restrict__ tok,
                        const float* __restrict__ pos, float* __restrict__ x) {
    int row = blockIdx.x;                    // b*T + t
    int t   = row & (Tc - 1);
    const float* te = tok + (size_t)idx[row] * Cc;
    const float* pe = pos + (size_t)t * Cc;
    float* xo = x + (size_t)row * Cc;
    for (int c = threadIdx.x; c < Cc; c += blockDim.x) xo[c] = te[c] + pe[c];
}

// ---------------------------------------------------------------- layernorm -> f16
__global__ void k_ln(const float* __restrict__ x, const float* __restrict__ g,
                     const float* __restrict__ b, half_t* __restrict__ out) {
    __shared__ float s1[256], s2[256];
    int row = blockIdx.x;
    const float* xr = x + (size_t)row * Cc;
    float sum = 0.f, sq = 0.f;
    for (int c = threadIdx.x; c < Cc; c += blockDim.x) {
        float v = xr[c]; sum += v; sq += v * v;
    }
    s1[threadIdx.x] = sum; s2[threadIdx.x] = sq;
    __syncthreads();
    for (int o = blockDim.x >> 1; o > 0; o >>= 1) {
        if (threadIdx.x < (unsigned)o) {
            s1[threadIdx.x] += s1[threadIdx.x + o];
            s2[threadIdx.x] += s2[threadIdx.x + o];
        }
        __syncthreads();
    }
    float mu  = s1[0] * (1.f / Cc);
    float var = s2[0] * (1.f / Cc) - mu * mu;
    float inv = rsqrtf(var + 1e-5f);
    half_t* po = out + (size_t)row * Cc;
    for (int c = threadIdx.x; c < Cc; c += blockDim.x)
        po[c] = (half_t)((xr[c] - mu) * inv * g[c] + b[c]);
}

// ------------------------------------------- weight convert: [K,N] f32 -> [N,K] f16
__global__ void k_transpose_w(const float* __restrict__ w, half_t* __restrict__ wt,
                              int K, int N) {
    size_t id = (size_t)blockIdx.x * blockDim.x + threadIdx.x;
    size_t total = (size_t)K * N;
    if (id >= total) return;
    size_t n = id / (size_t)K;
    size_t k = id - n * (size_t)K;
    wt[id] = (half_t)w[k * (size_t)N + n];
}

// ------------------- qkv weight convert: [H,C,HS] f32 -> [N=H*HS, K=C] f16 (transposed)
__global__ void k_conv_qkvw(const float* __restrict__ w, half_t* __restrict__ wt) {
    int id = blockIdx.x * blockDim.x + threadIdx.x;   // C*C = 1M
    if (id >= Cc * Cc) return;
    int n = id / Cc, c = id - n * Cc;
    int h = n >> 6, d = n & 63;
    wt[id] = (half_t)w[((size_t)h * Cc + c) * HSc + d];
}

// ---------------------------------------------------------------- WMMA GEMM
// C[M,N] = A[M,K](f16) * Bt[N,K](f16)^T + bias
// A tile (32x32 per K-step) is staged to LDS with async global->LDS copies
// (ASYNCcnt) and double-buffered; B fragments ping-pong in registers so the
// loads for k0+32 overlap the WMMAs for k0.
// mode 0: fp32 out (+optional resid add)  mode 1: f16 out (+optional relu)
// mode 2: f16 out transposed to [B,H,HS,T] (for V)
__global__ __launch_bounds__(256) void k_gemm(
    const half_t* __restrict__ A, const half_t* __restrict__ Bt,
    const float* __restrict__ bias, const float* __restrict__ resid,
    float* __restrict__ outF, half_t* __restrict__ outH,
    int M, int N, int K, int mode, int relu) {
    __shared__ alignas(16) half_t atile[2][32 * 32];   // 2 x 2KB

    int tid  = threadIdx.x;
    int lane = tid & 31;
    int wid  = tid >> 5;
    int wm = wid & 1, wn = wid >> 1;
    int hi = lane >> 4, nl = lane & 15;
    int mblock = blockIdx.y * 32;
    int nbase  = blockIdx.x * 256 + wn * 64;

    // async copy of one 32x32 f16 A tile: threads 0..127, 16 bytes each
    auto copyA = [&](int buf, int k0) {
        if (tid < 128) {
            int row = tid >> 2;
            int col = (tid & 3) * 8;
            const half_t* src = A + (size_t)(mblock + row) * K + k0 + col;
            unsigned dst = (unsigned)(uintptr_t)&atile[buf][row * 32 + col];
            asm volatile("global_load_async_to_lds_b128 %0, %1, off"
                         :: "v"(dst), "v"((unsigned long long)(uintptr_t)src)
                         : "memory");
        }
    };
    auto loadB = [&](frag* bf, int k0) {
        #pragma unroll
        for (int ft = 0; ft < 4; ft++) {
            const half_t* brow = Bt + (size_t)(nbase + ft * 16 + nl) * K + k0 + hi * 16;
            bf[ft].h[0] = *(const v8h*)brow;
            bf[ft].h[1] = *(const v8h*)(brow + 8);
        }
    };

    v8f z = {};
    v8f acc[4] = {z, z, z, z};

    auto mm = [&](frag* bf, int buf) {
        const half_t* ab = &atile[buf][(wm * 16 + nl) * 32];
        frag a;
        a.h[0] = *(const v8h*)(ab + hi * 16);       // ds_load_b128
        a.h[1] = *(const v8h*)(ab + hi * 16 + 8);   // ds_load_b128
        #pragma unroll
        for (int ft = 0; ft < 4; ft++)
            acc[ft] = __builtin_amdgcn_wmma_f32_16x16x32_f16(
                false, a.v, false, bf[ft].v, (short)0, acc[ft], false, false);
    };

    frag b0[4], b1[4];
    copyA(0, 0);
    loadB(b0, 0);
    asm volatile("s_wait_asynccnt 0x0" ::: "memory");
    WG_BARRIER();

    for (int k0 = 0; k0 < K; k0 += 64) {            // K is a multiple of 64
        // half-step A: consume buf0/b0, prefetch buf1/b1 for k0+32
        copyA(1, k0 + 32);
        loadB(b1, k0 + 32);
        mm(b0, 0);
        asm volatile("s_wait_asynccnt 0x0" ::: "memory");
        WG_BARRIER();
        // half-step B: consume buf1/b1, prefetch buf0/b0 for k0+64
        bool more = (k0 + 64 < K);
        if (more) {
            copyA(0, k0 + 64);
            loadB(b0, k0 + 64);
        }
        mm(b1, 1);
        if (more) {
            asm volatile("s_wait_asynccnt 0x0" ::: "memory");
            WG_BARRIER();
        }
    }

    #pragma unroll
    for (int ft = 0; ft < 4; ft++) {
        int col = nbase + ft * 16 + nl;
        float bv = bias ? bias[col] : 0.f;
        #pragma unroll
        for (int r = 0; r < 8; r++) {
            int row = mblock + wm * 16 + r + hi * 8;
            float v = acc[ft][r] + bv;
            if (relu)  v = fmaxf(v, 0.f);
            if (resid) v += resid[(size_t)row * N + col];
            if (mode == 0) {
                outF[(size_t)row * N + col] = v;
            } else if (mode == 1) {
                outH[(size_t)row * N + col] = (half_t)v;
            } else {  // vT store: [B,H,HS,T]
                int b_ = row >> 10, t_ = row & (Tc - 1);
                int h_ = col >> 6,  d_ = col & 63;
                outH[(((size_t)b_ * Hc + h_) * HSc + d_) * Tc + t_] = (half_t)v;
            }
        }
    }
}

// ---------------------------------------------------------------- flash attention
// q16,k16: [B,T,H*HS] f16 ; vT16: [B,H,HS,T] f16 ; o16: [B,T,H*HS] f16
__global__ __launch_bounds__(128) void k_flash(
    const half_t* __restrict__ q16, const half_t* __restrict__ k16,
    const half_t* __restrict__ vT16, half_t* __restrict__ o16) {
    __shared__ alignas(16) half_t plds[4 * 16 * 32];

    int lane = threadIdx.x & 31;
    int wave = threadIdx.x >> 5;
    int hi = lane >> 4, nl = lane & 15, ml = lane & 15;
    int qt = blockIdx.x * 4 + wave;        // query tile index (16 rows)
    int hh = blockIdx.y;
    int bb = blockIdx.z;
    int qbase = qt * 16;

    // Q fragments (16 x 64 over two K=32 steps), reused across all key blocks
    const half_t* qrow = q16 + ((size_t)(bb * Tc + qbase + ml)) * Cc + hh * HSc;
    frag qa[2];
    #pragma unroll
    for (int kk = 0; kk < 2; kk++) {
        qa[kk].h[0] = *(const v8h*)(qrow + kk * 32 + hi * 16);
        qa[kk].h[1] = *(const v8h*)(qrow + kk * 32 + hi * 16 + 8);
    }

    v8f z = {};
    v8f acc[4] = {z, z, z, z};
    float m_i[8], l_i[8];
    #pragma unroll
    for (int r = 0; r < 8; r++) { m_i[r] = -1e30f; l_i[r] = 0.f; }

    half_t* pw = plds + wave * (16 * 32);
    const half_t* vbase = vT16 + ((size_t)bb * Hc + hh) * HSc * (size_t)Tc;
    int m_hi = qbase + 15;

    for (int s0 = 0; s0 <= m_hi; s0 += 32) {
        float val[2][8];
        #pragma unroll
        for (int t2 = 0; t2 < 2; t2++) {
            int key_base = s0 + t2 * 16;
            int key = key_base + nl;
            const half_t* krow = k16 + ((size_t)(bb * Tc + key)) * Cc + hh * HSc;
            v8f s = {};
            #pragma unroll
            for (int kk = 0; kk < 2; kk++) {
                frag kb;
                kb.h[0] = *(const v8h*)(krow + kk * 32 + hi * 16);
                kb.h[1] = *(const v8h*)(krow + kk * 32 + hi * 16 + 8);
                s = __builtin_amdgcn_wmma_f32_16x16x32_f16(
                    false, qa[kk].v, false, kb.v, (short)0, s, false, false);
            }
            #pragma unroll
            for (int r = 0; r < 8; r++) {
                int mrow = qbase + r + hi * 8;
                val[t2][r] = (key <= mrow) ? s[r] * SCALE_F : -1e30f;
            }
        }
        // online softmax update (rows live on 16-lane half-groups)
        #pragma unroll
        for (int r = 0; r < 8; r++) {
            float rm = fmaxf(val[0][r], val[1][r]);
            rm = fmaxf(rm, __shfl_xor(rm, 1, 32));
            rm = fmaxf(rm, __shfl_xor(rm, 2, 32));
            rm = fmaxf(rm, __shfl_xor(rm, 4, 32));
            rm = fmaxf(rm, __shfl_xor(rm, 8, 32));
            float mnew = fmaxf(m_i[r], rm);
            float alpha = __expf(m_i[r] - mnew);
            float p0 = __expf(val[0][r] - mnew);
            float p1 = __expf(val[1][r] - mnew);
            val[0][r] = p0; val[1][r] = p1;
            float ps = p0 + p1;
            ps += __shfl_xor(ps, 1, 32);
            ps += __shfl_xor(ps, 2, 32);
            ps += __shfl_xor(ps, 4, 32);
            ps += __shfl_xor(ps, 8, 32);
            l_i[r] = l_i[r] * alpha + ps;
            m_i[r] = mnew;
            #pragma unroll
            for (int ft = 0; ft < 4; ft++) acc[ft][r] *= alpha;
        }
        // P (C-layout) -> LDS -> A-layout fragment
        #pragma unroll
        for (int t2 = 0; t2 < 2; t2++)
            #pragma unroll
            for (int r = 0; r < 8; r++)
                pw[(r + hi * 8) * 32 + t2 * 16 + nl] = (half_t)val[t2][r];
        asm volatile("s_wait_dscnt 0x0" ::: "memory");
        frag pa;
        pa.h[0] = *(const v8h*)(pw + ml * 32 + hi * 16);
        pa.h[1] = *(const v8h*)(pw + ml * 32 + hi * 16 + 8);
        // O += P * V  (V B-frags contiguous in transposed layout)
        #pragma unroll
        for (int ft = 0; ft < 4; ft++) {
            int d = ft * 16 + nl;
            const half_t* vp = vbase + (size_t)d * Tc + s0 + hi * 16;
            frag vb;
            vb.h[0] = *(const v8h*)vp;
            vb.h[1] = *(const v8h*)(vp + 8);
            acc[ft] = __builtin_amdgcn_wmma_f32_16x16x32_f16(
                false, pa.v, false, vb.v, (short)0, acc[ft], false, false);
        }
    }

    #pragma unroll
    for (int ft = 0; ft < 4; ft++)
        #pragma unroll
        for (int r = 0; r < 8; r++) {
            float o = acc[ft][r] / l_i[r];
            o16[((size_t)(bb * Tc + qbase + r + hi * 8)) * Cc + hh * HSc + ft * 16 + nl]
                = (half_t)o;
        }
}

// ---------------------------------------------------------------- orchestration
extern "C" void kernel_launch(void* const* d_in, const int* in_sizes, int n_in,
                              void* d_out, int out_size, void* d_ws, size_t ws_size,
                              hipStream_t stream) {
    (void)in_sizes; (void)n_in; (void)out_size; (void)ws_size;
    const int*   idx  = (const int*)  d_in[0];
    const float* tok  = (const float*)d_in[1];
    const float* pos  = (const float*)d_in[2];
    const float* wq   = (const float*)d_in[3];
    const float* bq   = (const float*)d_in[4];
    const float* wk   = (const float*)d_in[5];
    const float* bk   = (const float*)d_in[6];
    const float* wv   = (const float*)d_in[7];
    const float* bv   = (const float*)d_in[8];
    const float* wo   = (const float*)d_in[9];
    const float* bo   = (const float*)d_in[10];
    const float* ln1g = (const float*)d_in[11];
    const float* ln1b = (const float*)d_in[12];
    const float* w1   = (const float*)d_in[13];
    const float* b1   = (const float*)d_in[14];
    const float* w2   = (const float*)d_in[15];
    const float* b2   = (const float*)d_in[16];
    const float* ln2g = (const float*)d_in[17];
    const float* ln2b = (const float*)d_in[18];
    const float* lnfg = (const float*)d_in[19];
    const float* lnfb = (const float*)d_in[20];
    const float* lmw  = (const float*)d_in[21];
    const float* lmb  = (const float*)d_in[22];

    char* ws = (char*)d_ws;
    size_t off = 0;
    auto alloc = [&](size_t bytes) -> void* {
        void* p = ws + off;
        off = (off + bytes + 255) & ~(size_t)255;
        return p;
    };
    float*  x    = (float*) alloc((size_t)MR * Cc * 4);
    half_t* h16  = (half_t*)alloc((size_t)MR * Cc * 2);
    half_t* q16  = (half_t*)alloc((size_t)MR * Cc * 2);
    half_t* k16  = (half_t*)alloc((size_t)MR * Cc * 2);
    half_t* vT16 = (half_t*)alloc((size_t)MR * Cc * 2);
    half_t* o16  = (half_t*)alloc((size_t)MR * Cc * 2);
    half_t* ff16 = (half_t*)alloc((size_t)MR * FFc * 2);
    half_t* wqT  = (half_t*)alloc((size_t)Cc * Cc * 2);
    half_t* wkT  = (half_t*)alloc((size_t)Cc * Cc * 2);
    half_t* wvT  = (half_t*)alloc((size_t)Cc * Cc * 2);
    half_t* woT  = (half_t*)alloc((size_t)Cc * Cc * 2);
    half_t* w1T  = (half_t*)alloc((size_t)Cc * FFc * 2);
    half_t* w2T  = (half_t*)alloc((size_t)FFc * Cc * 2);
    half_t* lmT  = (half_t*)alloc((size_t)Vc * Cc * 2);

    int cc_blocks = (Cc * Cc + 255) / 256;

    k_embed<<<MR, 256, 0, stream>>>(idx, tok, pos, x);

    for (int l = 0; l < Lc; l++) {
        size_t wOff  = (size_t)l * Cc * Cc;
        size_t bOff  = (size_t)l * Cc;
        size_t w1Off = (size_t)l * Cc * FFc;
        size_t b1Off = (size_t)l * FFc;

        k_ln<<<MR, 256, 0, stream>>>(x, ln1g + bOff, ln1b + bOff, h16);

        k_conv_qkvw<<<cc_blocks, 256, 0, stream>>>(wq + wOff, wqT);
        k_conv_qkvw<<<cc_blocks, 256, 0, stream>>>(wk + wOff, wkT);
        k_conv_qkvw<<<cc_blocks, 256, 0, stream>>>(wv + wOff, wvT);

        dim3 gC(Cc / 256, MR / 32);
        k_gemm<<<gC, 256, 0, stream>>>(h16, wqT, bq + bOff, nullptr, nullptr, q16,
                                       MR, Cc, Cc, 1, 0);
        k_gemm<<<gC, 256, 0, stream>>>(h16, wkT, bk + bOff, nullptr, nullptr, k16,
                                       MR, Cc, Cc, 1, 0);
        k_gemm<<<gC, 256, 0, stream>>>(h16, wvT, bv + bOff, nullptr, nullptr, vT16,
                                       MR, Cc, Cc, 2, 0);

        k_flash<<<dim3(Tc / 64, Hc, Bc), 128, 0, stream>>>(q16, k16, vT16, o16);

        k_transpose_w<<<cc_blocks, 256, 0, stream>>>(wo + wOff, woT, Cc, Cc);
        k_gemm<<<gC, 256, 0, stream>>>(o16, woT, bo + bOff, x, x, nullptr,
                                       MR, Cc, Cc, 0, 0);

        k_ln<<<MR, 256, 0, stream>>>(x, ln2g + bOff, ln2b + bOff, h16);

        int cf_blocks = (int)(((size_t)Cc * FFc + 255) / 256);
        k_transpose_w<<<cf_blocks, 256, 0, stream>>>(w1 + w1Off, w1T, Cc, FFc);
        dim3 gF(FFc / 256, MR / 32);
        k_gemm<<<gF, 256, 0, stream>>>(h16, w1T, b1 + b1Off, nullptr, nullptr, ff16,
                                       MR, FFc, Cc, 1, 1);
        k_transpose_w<<<cf_blocks, 256, 0, stream>>>(w2 + w1Off, w2T, FFc, Cc);
        k_gemm<<<gC, 256, 0, stream>>>(ff16, w2T, b2 + bOff, x, x, nullptr,
                                       MR, Cc, FFc, 0, 0);
    }

    k_ln<<<MR, 256, 0, stream>>>(x, lnfg, lnfb, h16);
    int vc_blocks = (int)(((size_t)Vc * Cc + 255) / 256);
    k_transpose_w<<<vc_blocks, 256, 0, stream>>>(lmw, lmT, Cc, Vc);
    dim3 gV(Vc / 256, MR / 32);
    k_gemm<<<gV, 256, 0, stream>>>(h16, lmT, lmb, nullptr, (float*)d_out, nullptr,
                                   MR, Vc, Cc, 0, 0);
}
